// SWEGNNLayer_14920716387062
// MI455X (gfx1250) — matrix operand order repaired and checked by reading, not compiled, not run on hardware
//
#include <hip/hip_runtime.h>
#include <hip/hip_bf16.h>

typedef __attribute__((ext_vector_type(16))) _Float16 v16h;
typedef __attribute__((ext_vector_type(8)))  float    v8f;

// ---------------------------------------------------------------------------
// Edge kernel: per wave, 16 edges/iteration, features on M / edges on N.
//   x_{l+1}^T = W_l^T * x_l^T  with weights preloaded as A-tiles in registers.
//
// Zero-shuffle layer chaining: the D-layout of layer l is repacked verbatim
// into the next B operand (tile0 -> halves 0..7, tile1 -> halves 8..15),
// which realizes B[K] = x[p(K)] with the lane-independent row permutation
//   p: [0..7]->[0..7], [8..15]->[16..23], [16..23]->[8..15], [24..31]->[24..31]
// compensated for free by loading the next layer's A as A[m,K] = W[p(K),m].
//
// Layer-1 input rows are likewise permuted so the gathers split across the
// two lane halves (q: K0,1=snd h/z ; K16,17=rcv h/z ; K18..20=attr), with A0
// loaded as A0[m,K] = W0[q(K),m].  Biases ride in the WMMA C operand
// (C[m,n] = b[m], a per-lane v8f).  Layer-3 A replicates eW2^T over rows so
// every lane owns msg[its edge]; lanes 0..15 scatter with one f32 atomic.
// ---------------------------------------------------------------------------
__global__ __launch_bounds__(256) void swegnn_edge_kernel(
    const float* __restrict__ nf,      // (N,2): h_t, zb interleaved
    const int*   __restrict__ ei,      // (2,E): snd row then rcv row
    const float* __restrict__ ea,      // (E,3)
    const float* __restrict__ eW0,     // (7,32)
    const float* __restrict__ eb0,     // (32)
    const float* __restrict__ eW1,     // (32,32)
    const float* __restrict__ eb1,     // (32)
    const float* __restrict__ eW2,     // (32,1)
    const float* __restrict__ eb2,     // (1)
    float* __restrict__ agg,           // (N) accumulated messages
    int E)
{
    const int lane = (int)(threadIdx.x & 31u);
    const int g    = lane >> 4;        // lane group (K-half selector)
    const int m    = lane & 15;        // row (A) / edge column (B/D)

    // ---- Preload weight A-tiles ------------------------------------------
    // A-layout half idx: K = 16*g + idx, row = 16*t + m
    v16h A0[2], A1[2], A3;
    #pragma unroll
    for (int t = 0; t < 2; ++t) {
        const int mg = 16 * t + m;
        #pragma unroll
        for (int idx = 0; idx < 16; ++idx) {
            const int k  = 16 * g + idx;
            // layer-1 input-row remap q(K)
            const int q  = (k < 2) ? k : ((k >= 16 && k <= 20) ? k - 14 : -1);
            // layers 2/3 row permutation p(K) (middle 8-blocks swapped)
            const int pk = (k < 8 || k >= 24) ? k : (k < 16 ? k + 8 : k - 8);
            A0[t][idx] = (_Float16)((q >= 0) ? eW0[q * 32 + mg] : 0.0f);
            A1[t][idx] = (_Float16)eW1[pk * 32 + mg];
        }
    }
    #pragma unroll
    for (int idx = 0; idx < 16; ++idx) {
        const int k  = 16 * g + idx;
        const int pk = (k < 8 || k >= 24) ? k : (k < 16 ? k + 8 : k - 8);
        A3[idx] = (_Float16)eW2[pk];               // replicated over all rows m
    }

    // ---- Bias matrices for the WMMA C operand: C[m,n] = b[m] --------------
    v8f C0[2], C1[2], C2;
    #pragma unroll
    for (int r = 0; r < 8; ++r) {
        C0[0][r] = eb0[r + 8 * g];
        C0[1][r] = eb0[16 + r + 8 * g];
        C1[0][r] = eb1[r + 8 * g];
        C1[1][r] = eb1[16 + r + 8 * g];
        C2[r]    = eb2[0];
    }
    const v16h z16 = {};

    const float2* __restrict__ nf2 = (const float2*)nf;

    const int ntiles = (E + 15) >> 4;
    const int wid    = (int)((blockIdx.x * blockDim.x + threadIdx.x) >> 5);
    const int nw     = (int)((gridDim.x * blockDim.x) >> 5);

    for (int tile = wid; tile < ntiles; tile += nw) {
        const int  e  = tile * 16 + m;
        const bool ev = (e < E);
        const int  ec = ev ? e : 0;            // clamped: loads stay unconditional

        // Both halves read the (same) index pair -> coalesced broadcast.
        const int snd = ei[ec];
        const int rcv = ei[E + ec];

        // Split gather: lower half fetches sender h/z, upper half receiver h/z.
        const int    nidx = g ? rcv : snd;
        const float2 hz   = nf2[nidx];

        // Upper half fetches the edge attributes (K rows 18..20).
        float a0 = 0.f, a1 = 0.f, a2 = 0.f;
        if (g) {
            a0 = ea[3 * ec];  a1 = ea[3 * ec + 1];  a2 = ea[3 * ec + 2];
            // prefetch next tile's attributes (clamped address, WGP scope)
            long pe = (long)3 * (e + 16 * nw);
            long pmax = (long)3 * (E - 1);
            __builtin_prefetch(&ea[pe < pmax ? pe : pmax], 0, 3);
        }

        v16h b0 = {};
        b0[0] = (_Float16)hz.x;   // K = 16g+0 : snd.h / rcv.h
        b0[1] = (_Float16)hz.y;   // K = 16g+1 : snd.z / rcv.z
        b0[2] = (_Float16)a0;     // K = 18 on g1, zero row K=2 on g0
        b0[3] = (_Float16)a1;
        b0[4] = (_Float16)a2;

        // ---- Layer 1: x1^T = relu(eW0^T * pair^T + eb0) -------------------
        v8f d0a = __builtin_amdgcn_wmma_f32_16x16x32_f16(false, A0[0], false, b0,
                                                         (short)0, C0[0], false, false);
        v8f d0b = __builtin_amdgcn_wmma_f32_16x16x32_f16(false, A0[1], false, b0,
                                                         (short)0, C0[1], false, false);

        // D -> B (row-permuted by p, compensated in A1): cvt/pack + packed relu
        v16h b1;
        #pragma unroll
        for (int r = 0; r < 8; ++r) {
            b1[r]     = (_Float16)d0a[r];
            b1[8 + r] = (_Float16)d0b[r];
        }
        b1 = __builtin_elementwise_max(b1, z16);   // v_pk_max_num_f16 x8

        // ---- Layer 2: x2^T = relu(eW1'^T * x1'^T + eb1) -------------------
        v8f d1a = __builtin_amdgcn_wmma_f32_16x16x32_f16(false, A1[0], false, b1,
                                                         (short)0, C1[0], false, false);
        v8f d1b = __builtin_amdgcn_wmma_f32_16x16x32_f16(false, A1[1], false, b1,
                                                         (short)0, C1[1], false, false);
        v16h b2;
        #pragma unroll
        for (int r = 0; r < 8; ++r) {
            b2[r]     = (_Float16)d1a[r];
            b2[8 + r] = (_Float16)d1b[r];
        }
        b2 = __builtin_elementwise_max(b2, z16);

        // ---- Layer 3: msg^T = eW2'^T * x2'^T + eb2  (D[m,n] = msg[n]) -----
        v8f d2 = __builtin_amdgcn_wmma_f32_16x16x32_f16(false, A3, false, b2,
                                                        (short)0, C2, false, false);

        // ---- Scatter: agg[rcv[e]] += msg, one lane per edge ---------------
        if (g == 0 && ev) {
            atomicAdd(&agg[rcv], d2[0]);
        }
    }
}

// ---------------------------------------------------------------------------
// Node kernel: ni=[h,zb,agg] -> relu(3x32)+32x1 -> h_next = relu(h + upd).
// 64 MFLOP total; scalar f32 is already bandwidth-trivial.
// ---------------------------------------------------------------------------
__global__ __launch_bounds__(256) void swegnn_node_kernel(
    const float* __restrict__ nf,
    const float* __restrict__ agg,
    const float* __restrict__ nW0,   // (3,32)
    const float* __restrict__ nb0,   // (32)
    const float* __restrict__ nW1,   // (32,1)
    const float* __restrict__ nb1,   // (1)
    float* __restrict__ out,
    int N)
{
    const int i = (int)(blockIdx.x * blockDim.x + threadIdx.x);
    if (i >= N) return;
    const float h = nf[2 * i];
    const float z = nf[2 * i + 1];
    const float a = agg[i];
    float upd = nb1[0];
    #pragma unroll
    for (int j = 0; j < 32; ++j) {
        float y = fmaf(h, nW0[j],
                  fmaf(z, nW0[32 + j],
                  fmaf(a, nW0[64 + j], nb0[j])));
        y = fmaxf(y, 0.0f);
        upd = fmaf(y, nW1[j], upd);
    }
    out[i] = fmaxf(h + upd, 0.0f);
}

extern "C" void kernel_launch(void* const* d_in, const int* in_sizes, int n_in,
                              void* d_out, int out_size, void* d_ws, size_t ws_size,
                              hipStream_t stream)
{
    const float* nf  = (const float*)d_in[0];
    const int*   ei  = (const int*)  d_in[1];
    const float* ea  = (const float*)d_in[2];
    const float* eW0 = (const float*)d_in[3];
    const float* eb0 = (const float*)d_in[4];
    const float* eW1 = (const float*)d_in[5];
    const float* eb1 = (const float*)d_in[6];
    const float* eW2 = (const float*)d_in[7];
    const float* eb2 = (const float*)d_in[8];
    const float* nW0 = (const float*)d_in[9];
    const float* nb0 = (const float*)d_in[10];
    const float* nW1 = (const float*)d_in[11];
    const float* nb1 = (const float*)d_in[12];

    const int N = in_sizes[0] / 2;   // node_features is (N,2)
    const int E = in_sizes[1] / 2;   // edge_index is (2,E)

    float* agg = (float*)d_ws;       // N floats of scratch for segment-sum
    hipMemsetAsync(agg, 0, (size_t)N * sizeof(float), stream);

    const int edge_blocks = 1024;    // 8192 wave32s, grid-stride over 16-edge tiles
    swegnn_edge_kernel<<<edge_blocks, 256, 0, stream>>>(
        nf, ei, ea, eW0, eb0, eW1, eb1, eW2, eb2, agg, E);

    swegnn_node_kernel<<<(N + 255) / 256, 256, 0, stream>>>(
        nf, agg, nW0, nb0, nW1, nb1, (float*)d_out, N);
}
